// Attention_8959301780132
// MI455X (gfx1250) — matrix operand order repaired and checked by reading, not compiled, not run on hardware
//
#include <hip/hip_runtime.h>
#include <math.h>

// MI455X / gfx1250 fused causal self-attention, B x (T=64, C=64).
// f16 WMMA (16x16x32) with f32 accumulation; one workgroup (4 wave32) per batch.
// Weights pre-packed to f16 (pitch-72 LDS image) once in d_ws by a setup kernel,
// then moved global->LDS with CDNA5 async-to-LDS instructions (ASYNCcnt).

typedef _Float16 half8 __attribute__((ext_vector_type(8)));
typedef _Float16 v16h  __attribute__((ext_vector_type(16)));
typedef float    v8f   __attribute__((ext_vector_type(8)));

#define LP 72                     // LDS row pitch in halves: 144B (16B aligned, 36 words -> conflict-free)
#define MAT_HALVES (64 * LP)      // 4608 halves per 64x64 matrix image
#define W_HALVES   (4 * MAT_HALVES)
#define WS_HALVES  (W_HALVES + 128)   // + 64 f32 bias (128 halves)
#define WS_BYTES   (WS_HALVES * 2)    // 37120 B
#define WS_CHUNKS  (WS_BYTES / 16)    // 2320 x b128

union Frag { half8 h8[2]; v16h v; };

// A-fragment (16x32 f16, MxK) per ISA 7.12.2:
//   lane L: m = L&15, kb = (L>=16)?8:0 ; halves = rows m, cols {kb..kb+7} u {kb+16..kb+23}
__device__ __forceinline__ v16h frag_a(const _Float16* base /* &mat[row0][kstep*32] */) {
    const int lane = threadIdx.x & 31;
    const int m  = lane & 15;
    const int kb = (lane >> 4) << 3;
    const _Float16* p = base + m * LP + kb;
    Frag f;
    f.h8[0] = *(const half8*)(p);
    f.h8[1] = *(const half8*)(p + 16);
    return f.v;
}

// B-fragment (32x16 f16, KxN), column-major access pattern:
//   lane L: n = L&15, kb16 = (L>=16)?8:0 ; halves = Bt[n][kb16..kb16+7] u Bt[n][kb16+16..kb16+23]
__device__ __forceinline__ v16h frag_b(const _Float16* base /* &matT[n0][kstep*32] */) {
    const int lane = threadIdx.x & 31;
    const int n    = lane & 15;
    const int kb16 = (lane >> 4) << 3;
    const _Float16* p = base + n * LP + kb16;
    Frag f;
    f.h8[0] = *(const half8*)(p);
    f.h8[1] = *(const half8*)(p + 16);
    return f.v;
}

__device__ __forceinline__ v8f wmma16(v16h a, v16h b, v8f c) {
    return __builtin_amdgcn_wmma_f32_16x16x32_f16(false, a, false, b, (short)0, c, false, false);
}

// C/D layout: VGPR r -> row ( (lane>=16)?8:0 ) + r, col lane&15
__device__ __forceinline__ void store_tile(_Float16* dst /* &mat[row0][col0] */, v8f acc) {
    const int lane = threadIdx.x & 31;
    const int n  = lane & 15;
    const int rb = (lane >> 4) << 3;
#pragma unroll
    for (int r = 0; r < 8; ++r)
        dst[(rb + r) * LP + n] = (_Float16)acc[r];
}

// 64x64 f32 global -> f16 LDS (pitch LP), float4 coalesced
__device__ __forceinline__ void cvt_store(_Float16* dst, const float* __restrict__ src) {
    const float4* s4 = (const float4*)src;
    for (int i = threadIdx.x; i < 1024; i += 128) {
        float4 v = s4[i];
        _Float16* d = dst + (i >> 4) * LP + ((i & 15) << 2);
        d[0] = (_Float16)v.x; d[1] = (_Float16)v.y;
        d[2] = (_Float16)v.z; d[3] = (_Float16)v.w;
    }
}

// CDNA5 async global->LDS 16B copy (tracked by ASYNCcnt, bypasses VGPRs)
__device__ __forceinline__ void async_copy16(unsigned lds_off, const void* gaddr) {
    asm volatile("global_load_async_to_lds_b128 %0, %1, off"
                 :: "v"(lds_off), "v"(gaddr) : "memory");
}

// ---- one-shot setup: pack Wk|Wq(*1/8)|Wv|Wp as f16 pitch-72 images + f32 bias into ws ----
__global__ __launch_bounds__(256) void pack_weights_kernel(
    const float* __restrict__ Wk, const float* __restrict__ Wq,
    const float* __restrict__ Wv, const float* __restrict__ Wp,
    const float* __restrict__ bp, _Float16* __restrict__ ws) {
    const int tid = blockIdx.x * 256 + threadIdx.x;   // 64 blocks * 256 = 16384 = 4*4096
    const int m   = tid >> 12;
    const int idx = tid & 4095;
    const float* src; float scale;
    switch (m) {
        case 0:  src = Wk; scale = 1.0f;   break;
        case 1:  src = Wq; scale = 0.125f; break;     // fold 1/sqrt(64) into Wq
        case 2:  src = Wv; scale = 1.0f;   break;
        default: src = Wp; scale = 1.0f;   break;
    }
    ws[m * MAT_HALVES + (idx >> 6) * LP + (idx & 63)] = (_Float16)(src[idx] * scale);
    if (tid < 64) ((float*)(ws + W_HALVES))[tid] = bp[tid];
}

__global__ __launch_bounds__(128) void attn64_kernel(
    const float* __restrict__ x, const _Float16* __restrict__ wsW,
    float* __restrict__ out) {

    __shared__ __align__(16) _Float16 sXP[MAT_HALVES];  // x, later P (probs)
    __shared__ __align__(16) _Float16 sQO[MAT_HALVES];  // q (pre-scaled), later O
    __shared__ __align__(16) _Float16 sK [MAT_HALVES];
    __shared__ __align__(16) _Float16 sVt[MAT_HALVES];  // V transposed: Vt[c][t]
    __shared__ __align__(16) _Float16 sW [WS_HALVES];   // Wk|Wq|Wv|Wp f16 + bias f32

    const _Float16* sWk = sW;
    const _Float16* sWq = sW + 1 * MAT_HALVES;
    const _Float16* sWv = sW + 2 * MAT_HALVES;
    const _Float16* sWp = sW + 3 * MAT_HALVES;
    const float*    sBp = (const float*)(sW + W_HALVES);

    const int b     = blockIdx.x;
    const int tid   = threadIdx.x;
    const int lane  = tid & 31;
    const int w     = tid >> 5;      // wave id 0..3 -> row strip
    const int n     = lane & 15;
    const int hh    = lane >> 4;
    const int strip = w * 16;

    // kick off async weight-image copy (global->LDS, no VGPR roundtrip) ...
    {
        const unsigned lds_base = (unsigned)(size_t)(&sW[0]);
        const char* g = (const char*)wsW;
        for (int i = tid; i < WS_CHUNKS; i += 128)
            async_copy16(lds_base + i * 16, g + i * 16);
    }
    // ... overlapped with x f32->f16 conversion through VGPRs
    cvt_store(sXP, x + (size_t)b * 4096);
    asm volatile("s_wait_asynccnt 0x0" ::: "memory");
    __syncthreads();

    // ---- Projections: K = x Wk^T, Q = x (Wq/8)^T, Vt = Wv x^T ----
    {
        v16h ax0 = frag_a(&sXP[strip * LP + 0]);
        v16h ax1 = frag_a(&sXP[strip * LP + 32]);
        v16h av0 = frag_a(&sWv[strip * LP + 0]);   // A = Wv rows (channels strip)
        v16h av1 = frag_a(&sWv[strip * LP + 32]);
#pragma unroll
        for (int nt = 0; nt < 4; ++nt) {
            v8f z = {};
            const _Float16* kb = &sWk[nt * 16 * LP];
            v8f aK = wmma16(ax0, frag_b(kb), z);
            aK     = wmma16(ax1, frag_b(kb + 32), aK);
            store_tile(&sK[strip * LP + nt * 16], aK);

            const _Float16* qb = &sWq[nt * 16 * LP];
            v8f aQ = wmma16(ax0, frag_b(qb), z);
            aQ     = wmma16(ax1, frag_b(qb + 32), aQ);
            store_tile(&sQO[strip * LP + nt * 16], aQ);

            const _Float16* xb = &sXP[nt * 16 * LP];              // B = x rows (t tile)
            v8f aV = wmma16(av0, frag_b(xb), z);
            aV     = wmma16(av1, frag_b(xb + 32), aV);
            store_tile(&sVt[strip * LP + nt * 16], aV);           // Vt[c][t]
        }
    }
    __syncthreads();

    // ---- S = Q K^T (causal: only tiles ts <= w), softmax -> P (reuse sXP) ----
    {
        v16h aq0 = frag_a(&sQO[strip * LP + 0]);
        v16h aq1 = frag_a(&sQO[strip * LP + 32]);
        v8f accS[4];
#pragma unroll
        for (int ts = 0; ts < 4; ++ts) { v8f z = {}; accS[ts] = z; }
        for (int ts = 0; ts <= w; ++ts) {
            const _Float16* kb = &sK[ts * 16 * LP];
            v8f z = {};
            v8f a = wmma16(aq0, frag_b(kb), z);
            accS[ts] = wmma16(aq1, frag_b(kb + 32), a);
        }
        const float NEG = -__builtin_inff();
#pragma unroll
        for (int r = 0; r < 8; ++r) {
            const int rr = (hh << 3) + r;   // row within strip; global row = strip+rr
            float vals[4];
#pragma unroll
            for (int ts = 0; ts < 4; ++ts) {
                const bool live = (ts < w) | ((ts == w) & (n <= rr));
                vals[ts] = live ? accS[ts][r] : NEG;
            }
            float mx = fmaxf(fmaxf(vals[0], vals[1]), fmaxf(vals[2], vals[3]));
#pragma unroll
            for (int off = 8; off >= 1; off >>= 1)
                mx = fmaxf(mx, __shfl_xor(mx, off, 32));
            float e[4], s = 0.0f;
#pragma unroll
            for (int ts = 0; ts < 4; ++ts) { e[ts] = __expf(vals[ts] - mx); s += e[ts]; }
#pragma unroll
            for (int off = 8; off >= 1; off >>= 1)
                s += __shfl_xor(s, off, 32);
            const float inv = 1.0f / s;
#pragma unroll
            for (int ts = 0; ts < 4; ++ts)
                sXP[(strip + rr) * LP + ts * 16 + n] = (_Float16)(e[ts] * inv);
        }
    }
    // No barrier: P rows are wave-private; sVt already synced above.

    // ---- O = P V  (B from Vt rows: contiguous K) -> reuse sQO ----
    {
        v16h ap0 = frag_a(&sXP[strip * LP + 0]);
        v16h ap1 = frag_a(&sXP[strip * LP + 32]);
#pragma unroll
        for (int ct = 0; ct < 4; ++ct) {
            const _Float16* vb = &sVt[ct * 16 * LP];
            v8f z = {};
            v8f aO = wmma16(ap0, frag_b(vb), z);
            aO     = wmma16(ap1, frag_b(vb + 32), aO);
            store_tile(&sQO[strip * LP + ct * 16], aO);
        }
    }

    // ---- Y = O Wp^T + bp -> global ----
    {
        v16h ao0 = frag_a(&sQO[strip * LP + 0]);
        v16h ao1 = frag_a(&sQO[strip * LP + 32]);
        float* og = out + (size_t)b * 4096;
#pragma unroll
        for (int nt = 0; nt < 4; ++nt) {
            const _Float16* pb = &sWp[nt * 16 * LP];
            v8f z = {};
            v8f aY = wmma16(ao0, frag_b(pb), z);
            aY     = wmma16(ao1, frag_b(pb + 32), aY);
            const float bv = sBp[nt * 16 + n];
#pragma unroll
            for (int r = 0; r < 8; ++r)
                og[(strip + (hh << 3) + r) * 64 + nt * 16 + n] = aY[r] + bv;
        }
    }
}

extern "C" void kernel_launch(void* const* d_in, const int* in_sizes, int n_in,
                              void* d_out, int out_size, void* d_ws, size_t ws_size,
                              hipStream_t stream) {
    (void)n_in; (void)out_size; (void)ws_size;
    const float* x  = (const float*)d_in[0];
    const float* Wk = (const float*)d_in[1];
    const float* Wq = (const float*)d_in[2];
    const float* Wv = (const float*)d_in[3];
    const float* Wp = (const float*)d_in[4];
    const float* bp = (const float*)d_in[5];
    float* out = (float*)d_out;
    _Float16* ws = (_Float16*)d_ws;
    const int B = in_sizes[0] / (64 * 64);

    pack_weights_kernel<<<64, 256, 0, stream>>>(Wk, Wq, Wv, Wp, bp, ws);
    attn64_kernel<<<B, 128, 0, stream>>>(x, ws, out);
}
